// MultiHeadSelfAttention_18004502904867
// MI455X (gfx1250) — compile-verified, matrix operand order checked by reading, not run on hardware
//
#include <hip/hip_runtime.h>
#include <hip/hip_bf16.h>

// ---------------------------------------------------------------------------
// Multi-head self-attention, fully WMMA-based for gfx1250 (MI455X).
//   x:[B,C,E] f32, Wk/Wq/Wv:[H*D,E] f32, Wo:[E,H*D] f32 -> out:[B,C,E] f32
//   B=4, C=2048, E=1024, H=16, D=64
// Pipeline: f32->f16 cvt, 3x NT-GEMM (QKV, f16 out in [B,H,C,D], LDS-staged
// via global_load_async_to_lds_b128), flash-attention (online softmax,
// causal), NT-GEMM (f32 out).
// ---------------------------------------------------------------------------

typedef _Float16 v16h __attribute__((ext_vector_type(16)));
typedef float    v8f  __attribute__((ext_vector_type(8)));

#define B_BATCH 4
#define C_SEQ   2048
#define E_DIM   1024
#define H_NUM   16
#define D_HEAD  64

__device__ inline v8f wmma_f16(v16h a, v16h b, v8f c) {
    // v_wmma_f32_16x16x32_f16 : D = A(16x32 f16) x B(32x16 f16) + C(16x16 f32)
    return __builtin_amdgcn_wmma_f32_16x16x32_f16(
        /*neg_a=*/false, a, /*neg_b=*/false, b,
        /*c_mod=*/(short)0, c, /*reuse_a=*/false, /*reuse_b=*/false);
}

union FragU { uint4 u[2]; v16h v; };

// Load a 16x32 f16 A/B fragment row slice: lane holds row/col (lane&15),
// elements 0..7 = K kb+0..7, elements 8..15 = K kb+16..23 (kb = 0 or 8 by
// lane half). Source is K-contiguous, 16B aligned -> two b128 loads
// (global_load_b128 or ds_load_b128 depending on provenance).
__device__ inline v16h load_frag_contig(const _Float16* p, int kb) {
    FragU f;
    f.u[0] = *reinterpret_cast<const uint4*>(p + kb);
    f.u[1] = *reinterpret_cast<const uint4*>(p + 16 + kb);
    return f.v;
}

// Async 16-byte global->LDS copy (CDNA5): tracked with ASYNCcnt.
// dst_lds = LDS byte offset (low 32 bits of a generic pointer into __shared__).
__device__ inline void async_copy_b128(unsigned dst_lds, const void* src) {
    asm volatile("global_load_async_to_lds_b128 %0, %1, off"
                 :: "v"(dst_lds), "v"(src) : "memory");
}
__device__ inline void wait_async0() {
    asm volatile("s_wait_asynccnt 0" ::: "memory");
}

// ---------------------------------------------------------------------------
__global__ __launch_bounds__(256) void cvt_f32_f16_kernel(
    const float* __restrict__ in, _Float16* __restrict__ out, int n) {
    int i = blockIdx.x * blockDim.x + threadIdx.x;
    int stride = gridDim.x * blockDim.x;
    for (; i < n; i += stride) out[i] = (_Float16)in[i];
}

// ---------------------------------------------------------------------------
// C[m,n] = sum_k A[m,k]*Bw[n,k]   (A row-major MxK, Bw row-major NxK)
// mode 0: write f16 scattered into [B,H,C,D] (QKV);  mode 1: f32 row-major.
// M = 8192, N = K = 1024. WG tile 128x64, 8 waves, wave tile 32x32.
// K-slices staged in LDS with async copies: A slice fetched once (was 2x),
// B slice once (was 4x) per workgroup.
__global__ __launch_bounds__(256) void gemm_nt_kernel(
    const _Float16* __restrict__ A, const _Float16* __restrict__ Bw,
    float* __restrict__ outF, _Float16* __restrict__ outQKV, int mode) {
    __shared__ __align__(16) _Float16 Atile[128][32];   // 8 KB
    __shared__ __align__(16) _Float16 Btile[64][32];    // 4 KB

    const int K = E_DIM, N = E_DIM;
    const int tid  = threadIdx.x;
    const int lane = tid & 31;
    const int w    = tid >> 5;
    const int wm = w & 3, wn = w >> 2;            // 4x2 wave grid
    const int mblk = blockIdx.x * 128;
    const int nblk = blockIdx.y * 64;
    const int ln = lane & 15;
    const int kb = (lane >= 16) ? 8 : 0;

    v8f acc[2][2] = {};

    for (int ks = 0; ks < K; ks += 32) {
        __syncthreads();                       // previous tile fully consumed
        // ---- async fill: A 128x32 (512 x 16B), B 64x32 (256 x 16B) ----
#pragma unroll
        for (int i = 0; i < 2; ++i) {
            int idx = tid + i * 256;
            int row = idx >> 2, ch = idx & 3;  // 4 x 16B chunks per row
            const _Float16* src = A + (size_t)(mblk + row) * K + ks + ch * 8;
            async_copy_b128((unsigned)(uintptr_t)&Atile[row][ch * 8], src);
        }
        {
            int row = tid >> 2, ch = tid & 3;
            const _Float16* src = Bw + (size_t)(nblk + row) * K + ks + ch * 8;
            async_copy_b128((unsigned)(uintptr_t)&Btile[row][ch * 8], src);
        }
        wait_async0();                         // this wave's ASYNCcnt == 0
        __syncthreads();                       // tiles visible to all waves

        // ---- fragments from LDS (ds_load_b128) + 4 WMMAs ----
        const _Float16* Ar = &Atile[wm * 32 + ln][0];
        const _Float16* Br = &Btile[wn * 32 + ln][0];
        v16h a0 = load_frag_contig(Ar, kb);
        v16h a1 = load_frag_contig(Ar + 16 * 32, kb);
        v16h b0 = load_frag_contig(Br, kb);
        v16h b1 = load_frag_contig(Br + 16 * 32, kb);
        acc[0][0] = wmma_f16(a0, b0, acc[0][0]);
        acc[0][1] = wmma_f16(a0, b1, acc[0][1]);
        acc[1][0] = wmma_f16(a1, b0, acc[1][0]);
        acc[1][1] = wmma_f16(a1, b1, acc[1][1]);
    }

    const int rowadd = (lane >= 16) ? 8 : 0;  // C-layout: VGPR r -> M=r(+8)
#pragma unroll
    for (int i = 0; i < 2; ++i)
#pragma unroll
        for (int j = 0; j < 2; ++j)
#pragma unroll
            for (int r = 0; r < 8; ++r) {
                int m = mblk + wm * 32 + i * 16 + r + rowadd;
                int n = nblk + wn * 32 + j * 16 + ln;
                float val = acc[i][j][r];
                if (mode == 0) {
                    int b = m >> 11, c = m & (C_SEQ - 1);
                    int h = n >> 6,  d = n & (D_HEAD - 1);
                    outQKV[(((size_t)b * H_NUM + h) * C_SEQ + c) * D_HEAD + d] =
                        (_Float16)val;
                } else {
                    outF[(size_t)m * N + n] = val;
                }
            }
}

// ---------------------------------------------------------------------------
// Flash attention, causal. One block = (b,h, 128 queries); one wave = 16
// queries, loops over 32-key blocks up to the diagonal with online softmax.
__global__ __launch_bounds__(256) void flash_attn_kernel(
    const _Float16* __restrict__ Q, const _Float16* __restrict__ Kt,
    const _Float16* __restrict__ V, _Float16* __restrict__ attnOut) {
    __shared__ __align__(16) _Float16 Pbuf[8][16][32]; // per-wave P relayout

    const int blk  = blockIdx.x;
    const int qblk = blk & 15;             // C/128 = 16
    const int bh   = blk >> 4;             // b*H + h
    const int b    = bh >> 4;
    const int h    = bh & 15;
    const int tid  = threadIdx.x;
    const int lane = tid & 31;
    const int w    = tid >> 5;
    const int ln   = lane & 15;
    const int kb   = (lane >= 16) ? 8 : 0; // K half-offset AND C-layout row add
    const int qbase = qblk * 128 + w * 16;
    const float scale = 0.125f;            // 1/sqrt(64)
    const float NEG_INF = -__builtin_inff();

    const _Float16* Qb = Q  + (size_t)bh * C_SEQ * D_HEAD;
    const _Float16* Kb = Kt + (size_t)bh * C_SEQ * D_HEAD;
    const _Float16* Vb = V  + (size_t)bh * C_SEQ * D_HEAD;

    // Q fragments for this wave's 16 rows (K=64 -> two 32-chunks), loaded once.
    const _Float16* Qp = Qb + (size_t)(qbase + ln) * D_HEAD;
    v16h qa0 = load_frag_contig(Qp, kb);        // d 0..31
    v16h qa1 = load_frag_contig(Qp + 32, kb);   // d 32..63

    v8f o[4] = {};
    float m_run[8], l_run[8];
#pragma unroll
    for (int r = 0; r < 8; ++r) { m_run[r] = NEG_INF; l_run[r] = 0.0f; }

    const int njb = (qbase + 15) / 32 + 1;      // key blocks up to diagonal
    for (int jb = 0; jb < njb; ++jb) {
        const int n0 = jb * 32;

        // ---- S = Q x K^T (16x32 tile as two 16x16 accumulators) ----
        const _Float16* Kp0 = Kb + (size_t)(n0 + ln) * D_HEAD;
        const _Float16* Kp1 = Kb + (size_t)(n0 + 16 + ln) * D_HEAD;
        v16h k00 = load_frag_contig(Kp0, kb);
        v16h k01 = load_frag_contig(Kp0 + 32, kb);
        v16h k10 = load_frag_contig(Kp1, kb);
        v16h k11 = load_frag_contig(Kp1 + 32, kb);
        v8f s0 = {}, s1 = {};
        s0 = wmma_f16(qa0, k00, s0);
        s0 = wmma_f16(qa1, k01, s0);
        s1 = wmma_f16(qa0, k10, s1);
        s1 = wmma_f16(qa1, k11, s1);

        // ---- scale + causal mask + per-row block max ----
        float bmax[8];
#pragma unroll
        for (int r = 0; r < 8; ++r) {
            int mg = qbase + r + kb;           // global query row
            float v0 = s0[r] * scale;
            float v1 = s1[r] * scale;
            if (n0 + ln      > mg) v0 = NEG_INF;
            if (n0 + 16 + ln > mg) v1 = NEG_INF;
            s0[r] = v0; s1[r] = v1;
            bmax[r] = fmaxf(v0, v1);
        }
#pragma unroll
        for (int r = 0; r < 8; ++r) {          // reduce across 16-lane half
            bmax[r] = fmaxf(bmax[r], __shfl_xor(bmax[r], 1, 32));
            bmax[r] = fmaxf(bmax[r], __shfl_xor(bmax[r], 2, 32));
            bmax[r] = fmaxf(bmax[r], __shfl_xor(bmax[r], 4, 32));
            bmax[r] = fmaxf(bmax[r], __shfl_xor(bmax[r], 8, 32));
        }

        // ---- online softmax update, P -> LDS (f16) ----
#pragma unroll
        for (int r = 0; r < 8; ++r) {
            float mnew  = fmaxf(m_run[r], bmax[r]);
            float alpha = __expf(m_run[r] - mnew);
            m_run[r] = mnew;
            float p0 = __expf(s0[r] - mnew);
            float p1 = __expf(s1[r] - mnew);
            float rs = p0 + p1;
            rs += __shfl_xor(rs, 1, 32);
            rs += __shfl_xor(rs, 2, 32);
            rs += __shfl_xor(rs, 4, 32);
            rs += __shfl_xor(rs, 8, 32);
            l_run[r] = l_run[r] * alpha + rs;
#pragma unroll
            for (int t = 0; t < 4; ++t) o[t][r] *= alpha;
            int row = r + kb;                  // C-layout row within tile
            Pbuf[w][row][ln]      = (_Float16)p0;
            Pbuf[w][row][ln + 16] = (_Float16)p1;
        }

        // ---- P(16x32) x V(32x64): A-fragment from LDS, 4 d-tiles ----
        v16h pf;
#pragma unroll
        for (int j = 0; j < 8; ++j) {
            pf[j]     = Pbuf[w][ln][kb + j];
            pf[j + 8] = Pbuf[w][ln][16 + kb + j];
        }
#pragma unroll
        for (int t = 0; t < 4; ++t) {
            int dcol = t * 16 + ln;
            v16h vf;
#pragma unroll
            for (int j = 0; j < 8; ++j) {
                vf[j]     = Vb[(size_t)(n0 + kb + j) * D_HEAD + dcol];
                vf[j + 8] = Vb[(size_t)(n0 + 16 + kb + j) * D_HEAD + dcol];
            }
            o[t] = wmma_f16(pf, vf, o[t]);
        }
    }

    // ---- normalize and write [B, C, H*D] f16 ----
#pragma unroll
    for (int t = 0; t < 4; ++t)
#pragma unroll
        for (int r = 0; r < 8; ++r) {
            int row = qbase + r + kb;
            float val = o[t][r] / l_run[r];
            attnOut[((size_t)b * C_SEQ + row) * (H_NUM * D_HEAD) +
                    h * D_HEAD + t * 16 + ln] = (_Float16)val;
        }
}

// ---------------------------------------------------------------------------
extern "C" void kernel_launch(void* const* d_in, const int* in_sizes, int n_in,
                              void* d_out, int out_size, void* d_ws, size_t ws_size,
                              hipStream_t stream) {
    const float* x  = (const float*)d_in[0];
    const float* Wk = (const float*)d_in[1];
    const float* Wq = (const float*)d_in[2];
    const float* Wv = (const float*)d_in[3];
    const float* Wo = (const float*)d_in[4];
    float* out = (float*)d_out;

    const size_t NX = (size_t)B_BATCH * C_SEQ * E_DIM;   // 8388608
    const size_t NW = (size_t)E_DIM * E_DIM;             // 1048576

    char* ws = (char*)d_ws;
    size_t off = 0;
    _Float16* x16  = (_Float16*)(ws + off); off += NX * 2;
    _Float16* wq16 = (_Float16*)(ws + off); off += NW * 2;
    _Float16* wk16 = (_Float16*)(ws + off); off += NW * 2;
    _Float16* wv16 = (_Float16*)(ws + off); off += NW * 2;
    _Float16* wo16 = (_Float16*)(ws + off); off += NW * 2;
    _Float16* Q16  = (_Float16*)(ws + off); off += NX * 2;  // [B,H,C,D]
    _Float16* K16  = (_Float16*)(ws + off); off += NX * 2;
    _Float16* V16  = (_Float16*)(ws + off); off += NX * 2;
    _Float16* A16  = (_Float16*)(ws + off); off += NX * 2;  // [B,C,H*D]

    // f32 -> f16 conversions
    cvt_f32_f16_kernel<<<2048, 256, 0, stream>>>(x,  x16,  (int)NX);
    cvt_f32_f16_kernel<<<512,  256, 0, stream>>>(Wq, wq16, (int)NW);
    cvt_f32_f16_kernel<<<512,  256, 0, stream>>>(Wk, wk16, (int)NW);
    cvt_f32_f16_kernel<<<512,  256, 0, stream>>>(Wv, wv16, (int)NW);
    cvt_f32_f16_kernel<<<512,  256, 0, stream>>>(Wo, wo16, (int)NW);

    // QKV projections: [8192,1024] x [1024,1024]^T -> [B,H,C,D] f16
    dim3 ggrid(64, 16);   // M/128, N/64
    gemm_nt_kernel<<<ggrid, 256, 0, stream>>>(x16, wq16, nullptr, Q16, 0);
    gemm_nt_kernel<<<ggrid, 256, 0, stream>>>(x16, wk16, nullptr, K16, 0);
    gemm_nt_kernel<<<ggrid, 256, 0, stream>>>(x16, wv16, nullptr, V16, 0);

    // Flash attention: B*H * (C/128) = 1024 blocks
    flash_attn_kernel<<<B_BATCH * H_NUM * (C_SEQ / 128), 256, 0, stream>>>(
        Q16, K16, V16, A16);

    // Output projection: [8192,1024] x Wo[1024,1024]^T -> f32 out
    gemm_nt_kernel<<<ggrid, 256, 0, stream>>>(A16, wo16, out, nullptr, 1);
}
